// CRFLoss_32676111188394
// MI455X (gfx1250) — matrix operand order repaired and checked by reading, not compiled, not run on hardware
//
#include <hip/hip_runtime.h>
#include <hip/hip_bf16.h>

// ---------------------------------------------------------------------------
// CRF mean-field loss on MI455X (gfx1250), wave32 + WMMA f16 16x16x32.
//
// Pipeline:
//   1) build_ksum : Ksum[i][j] = exp(-.5*d2g) + exp(-.5*d2b), diag=0, f16 (162MB, L2-resident)
//   2) unary_q0   : U = -log_softmax(logits), Q0 = softmax(logits) (class-major f16), CE via atomics
//   3) crf_step x5: msg = Ksum @ Q  (v_wmma_f32_16x16x32_f16), fused Potts + softmax epilogue
//   4) finalize   : out[c,i] = ce + Q[c][i]
// ---------------------------------------------------------------------------

typedef __attribute__((ext_vector_type(16))) _Float16 v16h;
typedef __attribute__((ext_vector_type(8)))  _Float16 v8h;
typedef __attribute__((ext_vector_type(8)))  float    v8f;

#define HH   96
#define WW   96
#define NPIX 9216            // HH*WW
#define NC   21
#define CPAD 32
#define COMPAT 10.0f
#define KSTEPS ((NPIX / 4) / 32)   // 72 WMMA k-steps per wave (K split 4 ways)

// ---------------------------------------------------------------------------
// Kernel 1: fused pairwise kernel matrix, f16.  grid=(NPIX/256, NPIX), block=256
// ---------------------------------------------------------------------------
__global__ void build_ksum_kernel(const float* __restrict__ image,
                                  _Float16* __restrict__ Ksum) {
    const int i = blockIdx.y;
    const int j = blockIdx.x * 256 + threadIdx.x;

    const int yi = i / WW, xi = i - yi * WW;
    const int yj = j / WW, xj = j - yj * WW;

    const float ri = image[i],            rj = image[j];
    const float gi = image[NPIX + i],     gj = image[NPIX + j];
    const float bi = image[2 * NPIX + i], bj = image[2 * NPIX + j];

    const float dy = (float)(yi - yj), dx = (float)(xi - xj);
    const float d2 = dy * dy + dx * dx;
    const float dr = ri - rj, dg = gi - gj, db = bi - bj;
    const float dc2 = dr * dr + dg * dg + db * db;

    // gaussian: pos / 64 ; bilateral: pos / 3, rgb / 255
    const float kg = __expf(-0.5f * d2 * (1.0f / (64.0f * 64.0f)));
    const float kb = __expf(-0.5f * (d2 * (1.0f / 9.0f) + dc2 * (1.0f / 65025.0f)));

    float v = (i == j) ? 0.0f : (kg + kb);   // both identities removed
    Ksum[(size_t)i * NPIX + j] = (_Float16)v;
}

// ---------------------------------------------------------------------------
// Kernel 2: unaries, Q0 (class-major f16, padded classes zeroed), CE reduction.
// grid = NPIX/256, block = 256 (one thread per pixel)
// ---------------------------------------------------------------------------
__global__ void unary_q0_kernel(const float* __restrict__ logits,
                                const int*   __restrict__ labels,
                                float*    __restrict__ Ut,     // [CPAD][NPIX], stores -logp
                                _Float16* __restrict__ QtA,    // [CPAD][NPIX]
                                _Float16* __restrict__ QtB,    // [CPAD][NPIX]
                                float*    __restrict__ ce_acc) {
    __shared__ float sred[256];
    const int i = blockIdx.x * 256 + threadIdx.x;

    float x[NC];
    float m = -3.4e38f;
    #pragma unroll
    for (int c = 0; c < NC; ++c) {
        x[c] = logits[c * NPIX + i];
        m = fmaxf(m, x[c]);
    }
    float s = 0.0f;
    #pragma unroll
    for (int c = 0; c < NC; ++c) s += __expf(x[c] - m);
    const float lse = __logf(s);

    #pragma unroll
    for (int c = 0; c < NC; ++c) {
        const float lp = x[c] - m - lse;           // log-prob
        Ut[c * NPIX + i]  = -lp;                   // U
        QtA[c * NPIX + i] = (_Float16)__expf(lp);  // Q0 = softmax
    }
    #pragma unroll
    for (int c = NC; c < CPAD; ++c) {              // padded classes contribute 0
        QtA[c * NPIX + i] = (_Float16)0.0f;
        QtB[c * NPIX + i] = (_Float16)0.0f;
    }

    // cross-entropy contribution: -logp[label] / N
    const int l = labels[i];
    float ce = -(x[l] - m - lse) * (1.0f / (float)NPIX);

    // block reduction, then one atomic per block
    sred[threadIdx.x] = ce;
    __syncthreads();
    for (int off = 128; off > 0; off >>= 1) {
        if (threadIdx.x < off) sred[threadIdx.x] += sred[threadIdx.x + off];
        __syncthreads();
    }
    if (threadIdx.x == 0) atomicAdd(ce_acc, sred[0]);
}

// ---------------------------------------------------------------------------
// Kernel 3: one mean-field step.
//   msg = Ksum @ Qin ; pair = COMPAT*(rowsum - msg) ; Qout = softmax(-U - pair)
// Block = 128 threads (4 waves); block owns 16 output rows; waves split K.
// grid = NPIX/16 = 576.  Constant trip-count loop + uniform-base/int-offset
// addressing -> scalar loop control + SADDR-form global loads.
// ---------------------------------------------------------------------------
__global__ void __launch_bounds__(128)
crf_step_kernel(const _Float16* __restrict__ Ksum,   // [NPIX][NPIX]
                const _Float16* __restrict__ Qin,    // [CPAD][NPIX] class-major
                const float*    __restrict__ Ut,     // [CPAD][NPIX]
                _Float16*       __restrict__ Qout) { // [CPAD][NPIX]
    __shared__ float red[4][16][33];                 // [wave][row][col] padded

    const int tid   = threadIdx.x;
    const int lane  = tid & 31;
    const int wave  = tid >> 5;
    const int row   = lane & 15;          // A-matrix row within tile / C col index
    const int khalf = lane >> 4;          // 0 or 1

    const int r0   = blockIdx.x * 16;     // output rows r0..r0+15
    const int kbeg = wave * (NPIX / 4);   // this wave's K chunk

    v8f acc0 = {};                        // classes  0..15
    v8f acc1 = {};                        // classes 16..31

    // 32-bit element offsets against uniform base pointers (GVS addressing)
    const int aoff  = (r0 + row) * NPIX + kbeg + khalf * 16;  // A: 16x32 tile
    const int b0off = row        * NPIX + kbeg + khalf * 16;  // B: classes 0..15
    const int b1off = (row + 16) * NPIX + kbeg + khalf * 16;  // B: classes 16..31

    #pragma unroll 4
    for (int t = 0; t < KSTEPS; ++t) {
        const int k0 = t * 32;

        // ---- A tile: elems 0-15 = K(khalf*16..khalf*16+15) contiguous
        v16h a;
        ((v8h*)&a)[0] = *(const v8h*)(Ksum + aoff + k0);
        ((v8h*)&a)[1] = *(const v8h*)(Ksum + aoff + k0 + 8);

        // prefetch A stream ahead (global_prefetch_b8); Ksum is L2-resident
        __builtin_prefetch(Ksum + aoff + k0 + 256, 0, 3);

        // ---- B tiles: lane holds 16 contiguous K values starting at khalf*16
        const v16h b0 = *(const v16h*)(Qin + b0off + k0);
        const v16h b1 = *(const v16h*)(Qin + b1off + k0);

        acc0 = __builtin_amdgcn_wmma_f32_16x16x32_f16(
            false, a, false, b0, (short)0, acc0, false, false);
        acc1 = __builtin_amdgcn_wmma_f32_16x16x32_f16(
            false, a, false, b1, (short)0, acc1, false, false);
    }

    // ---- scatter C tiles to LDS: C layout => row = v + 8*khalf, col = lane&15
    #pragma unroll
    for (int v = 0; v < 8; ++v) {
        red[wave][v + 8 * khalf][row]      = acc0[v];
        red[wave][v + 8 * khalf][row + 16] = acc1[v];
    }
    __syncthreads();

    // ---- epilogue: reduce 4 wave partials, Potts compat, softmax, write Qout
    if (tid < 16) {
        const int r = r0 + tid;
        float msg[NC];
        float s = 0.0f;
        #pragma unroll
        for (int c = 0; c < NC; ++c) {
            const float v = red[0][tid][c] + red[1][tid][c]
                          + red[2][tid][c] + red[3][tid][c];
            msg[c] = v;
            s += v;
        }
        float lg[NC];
        float mx = -3.4e38f;
        #pragma unroll
        for (int c = 0; c < NC; ++c) {
            const float pair = COMPAT * (s - msg[c]);
            lg[c] = -Ut[c * NPIX + r] - pair;
            mx = fmaxf(mx, lg[c]);
        }
        float Z = 0.0f;
        #pragma unroll
        for (int c = 0; c < NC; ++c) {
            const float e = __expf(lg[c] - mx);
            lg[c] = e;
            Z += e;
        }
        const float inv = 1.0f / Z;
        #pragma unroll
        for (int c = 0; c < NC; ++c)
            Qout[c * NPIX + r] = (_Float16)(lg[c] * inv);
    }
}

// ---------------------------------------------------------------------------
// Kernel 4: out[c*N + i] = ce + Q[c][i]
// ---------------------------------------------------------------------------
__global__ void finalize_kernel(const _Float16* __restrict__ Qfin,
                                const float* __restrict__ ce_acc,
                                float* __restrict__ out) {
    const int idx = blockIdx.x * 256 + threadIdx.x;
    if (idx >= NC * NPIX) return;
    const int c = idx / NPIX;
    const int i = idx - c * NPIX;
    out[idx] = ce_acc[0] + (float)Qfin[c * NPIX + i];
}

// ---------------------------------------------------------------------------
// Host launch
// ---------------------------------------------------------------------------
extern "C" void kernel_launch(void* const* d_in, const int* in_sizes, int n_in,
                              void* d_out, int out_size, void* d_ws, size_t ws_size,
                              hipStream_t stream) {
    const float* logits = (const float*)d_in[0];   // [1,21,96,96]
    const int*   labels = (const int*)  d_in[1];   // [1,96,96]
    const float* image  = (const float*)d_in[2];   // [1,3,96,96]
    float* out = (float*)d_out;                    // [1,21,96,96]

    // workspace carve-up
    char* ws = (char*)d_ws;
    const size_t K_BYTES  = (size_t)NPIX * NPIX * sizeof(_Float16);   // 162 MB
    const size_t U_BYTES  = (size_t)CPAD * NPIX * sizeof(float);      // 1.13 MB
    const size_t Q_BYTES  = (size_t)CPAD * NPIX * sizeof(_Float16);   // 576 KB

    _Float16* Ksum = (_Float16*)ws;
    float*    Ut   = (float*)   (ws + K_BYTES);
    _Float16* QtA  = (_Float16*)(ws + K_BYTES + U_BYTES);
    _Float16* QtB  = (_Float16*)(ws + K_BYTES + U_BYTES + Q_BYTES);
    float*    ce   = (float*)   (ws + K_BYTES + U_BYTES + 2 * Q_BYTES);

    hipMemsetAsync(ce, 0, sizeof(float), stream);

    // 1) pairwise kernel matrix
    build_ksum_kernel<<<dim3(NPIX / 256, NPIX), 256, 0, stream>>>(image, Ksum);

    // 2) unaries + Q0 + CE
    unary_q0_kernel<<<NPIX / 256, 256, 0, stream>>>(logits, labels, Ut, QtA, QtB, ce);

    // 3) 5 mean-field iterations, ping-pong Q buffers (A->B->A->B->A->B)
    _Float16* qin  = QtA;
    _Float16* qout = QtB;
    for (int it = 0; it < 5; ++it) {
        crf_step_kernel<<<NPIX / 16, 128, 0, stream>>>(Ksum, qin, Ut, qout);
        _Float16* t = qin; qin = qout; qout = t;
    }
    // after 5 iterations final Q is in `qin`

    // 4) broadcast-add CE
    finalize_kernel<<<(NC * NPIX + 255) / 256, 256, 0, stream>>>(qin, ce, out);
}
